// Symmetric_CA_Layer_618475290758
// MI455X (gfx1250) — compile-verified, matrix operand order checked by reading, not run on hardware
//
#include <hip/hip_runtime.h>

#define BATCH 4
#define CDIM  256
#define NDIM  4096
#define DDIM  64

typedef unsigned short u16;
typedef unsigned int   u32;
typedef __attribute__((ext_vector_type(16))) __bf16 v16bf;
typedef __attribute__((ext_vector_type(8)))  __bf16 v8bf;
typedef __attribute__((ext_vector_type(8)))  float  v8f;

// float -> bf16 bits, round-to-nearest-even
static __device__ __forceinline__ u16 bf_bits(float f) {
  unsigned u = __builtin_bit_cast(unsigned, f);
  return (u16)((u + 0x7FFFu + ((u >> 16) & 1u)) >> 16);
}
static __device__ __forceinline__ u32 bf_pack2(float a, float b) {
  return (u32)bf_bits(a) | ((u32)bf_bits(b) << 16);
}

// byte offset of a __shared__ object inside LDS (for async-load VDST operand)
static __device__ __forceinline__ unsigned lds_off(const void* p) {
  return (unsigned)(unsigned long long)(__attribute__((address_space(3))) const void*)p;
}

// gfx1250 async copy: 16 bytes global -> LDS per lane, tracked by ASYNCcnt
static __device__ __forceinline__ void async_copy_b128(unsigned lds_byte_off,
                                                       const void* gaddr) {
  asm volatile("global_load_async_to_lds_b128 %0, %1, off"
               :: "v"(lds_byte_off), "v"(gaddr)
               : "memory");
}
// threshold wait: async loads complete in order, so waiting for ASYNCcnt<=N
// guarantees everything issued before the newest N ops has landed
template <int N>
static __device__ __forceinline__ void wait_async() {
#if __has_builtin(__builtin_amdgcn_s_wait_asynccnt)
  __builtin_amdgcn_s_wait_asynccnt(N);
#else
  asm volatile("s_wait_asynccnt %0" :: "i"(N) : "memory");
#endif
}

// Load one 16x32 bf16 operand tile (row-major in LDS, pitch in u16 units) into
// the gfx1250 WMMA 16-bit per-lane layout:
//   lanes 0-15 : row = lane,    K 0-7 and 16-23
//   lanes 16-31: row = lane-16, K 8-15 and 24-31
static __device__ __forceinline__ v16bf load_frag(const u16* base, int pitch, int lane) {
  int r  = lane & 15;
  int kh = (lane >> 4) << 3;            // 0 or 8
  const u16* row = base + r * pitch + kh;
  v8bf lo = *(const v8bf*)(row);        // K = kh .. kh+7
  v8bf hi = *(const v8bf*)(row + 16);   // K = 16+kh .. 16+kh+7
  return __builtin_shufflevector(lo, hi, 0,1,2,3,4,5,6,7,8,9,10,11,12,13,14,15);
}

static __device__ __forceinline__ v8f wmma_bf16(v16bf a, v16bf b, v8f c) {
  return __builtin_amdgcn_wmma_f32_16x16x32_bf16(false, a, false, b, (short)0, c, false, false);
}

// ---------------------------------------------------------------------------
// Projection GEMM: Out[b,o,n] = sum_c W[o,c] * X[b,c,n]   (Out in bf16 bits)
// transOut=1 stores Out[b,n,o] (used for q AND k so attention tiles are
// row-major contiguous and can be async-staged).
// ---------------------------------------------------------------------------
__global__ void __launch_bounds__(128)
proj_gemm_kernel(const float* __restrict__ W, const float* __restrict__ X,
                 u16* __restrict__ Out, int O, int K, int transOut)
{
  const int b    = blockIdx.z;
  const int row0 = blockIdx.y * 64;
  const int col0 = blockIdx.x * 64;
  const int tid  = threadIdx.x;
  const int wave = tid >> 5, lane = tid & 31;

  __shared__ __align__(16) u16 lA [64 * 32];   // [o][k]
  __shared__ __align__(16) u16 lBT[64 * 32];   // [n][k] (transposed while staging)

  const float* Xb = X + (size_t)b * K * NDIM;

  v8f acc[4] = {};
  for (int k0 = 0; k0 < K; k0 += 32) {
    for (int i = tid; i < 64 * 16; i += 128) {       // A: packed u32 stores
      int o = i >> 4, kp = (i & 15) * 2;
      const float* wp = W + (size_t)(row0 + o) * K + (k0 + kp);
      ((u32*)lA)[i] = bf_pack2(wp[0], wp[1]);
    }
    for (int i = tid; i < 64 * 32; i += 128) {       // B: fp32->bf16 transpose
      int kk = i >> 6, n = i & 63;
      lBT[n * 32 + kk] = bf_bits(Xb[(size_t)(k0 + kk) * NDIM + (col0 + n)]);
    }
    __syncthreads();
    v16bf a = load_frag(lA + (wave * 16) * 32, 32, lane);
    #pragma unroll
    for (int ct = 0; ct < 4; ++ct) {
      v16bf bb = load_frag(lBT + (ct * 16) * 32, 32, lane);
      acc[ct] = wmma_bf16(a, bb, acc[ct]);
    }
    __syncthreads();
  }
  #pragma unroll
  for (int ct = 0; ct < 4; ++ct) {
    int n  = col0 + ct * 16 + (lane & 15);
    int ob = row0 + wave * 16 + ((lane >> 4) << 3);
    #pragma unroll
    for (int v = 0; v < 8; ++v) {
      int o = ob + v;
      size_t idx = transOut ? ((size_t)b * NDIM * O + (size_t)n * O + o)
                            : ((size_t)b * O * NDIM + (size_t)o * NDIM + n);
      Out[idx] = bf_bits(acc[ct][v]);
    }
  }
}

// ---------------------------------------------------------------------------
// Pipelined flash attention. Per (batch, 64-query tile), stream N in 64-tiles:
//   S = Q.K^T (WMMA, fp32 acc) -> online softmax (fp32) -> P (bf16)
//   O[c,m] += V.P^T (WMMA, fp32 acc) with running-max rescale.
// Depth-1 async pipeline with ASYNCcnt thresholds (in-order completion):
//   per wave per iter: KT[next] = 4 async ops, V[next] = 16 async ops.
//   - wait<=20/16 before S-WMMA  => KT[cur] landed (V[cur], KT[next] in flight)
//   - wait<=4/0  before O-WMMA   => V[cur] landed  (KT[next] in flight)
// r is written transposed [B][N][C] with 16B packed stores.
// ---------------------------------------------------------------------------
__global__ void __launch_bounds__(128)
fmha_kernel(const u16* __restrict__ Qt,   // [B][N][D] bf16
            const u16* __restrict__ Kt,   // [B][N][D] bf16 (pre-transposed)
            const u16* __restrict__ Vm,   // [B][C][N] bf16
            u16* __restrict__ Rt)         // [B][N][C] bf16 (transposed out)
{
  const int b   = blockIdx.y;
  const int m0  = blockIdx.x * 64;
  const int tid = threadIdx.x;
  const int wave = tid >> 5, lane = tid & 31;

  __shared__ __align__(16) u16   sQ  [64 * DDIM];   //  8 KB [m][d]
  __shared__ __align__(16) u16   sKT0[64 * DDIM];   //  8 KB [n][d]
  __shared__ __align__(16) u16   sKT1[64 * DDIM];   //  8 KB [n][d]
  __shared__ __align__(16) float sS  [64 * 64];     // 16 KB [m][n]
  __shared__ __align__(16) u16   sP  [64 * 64];     //  8 KB [m][n]
  __shared__ __align__(16) u16   sV  [CDIM * 64];   // 32 KB [c][n]
  __shared__ float sMax[64], sSum[64], sScale[64], sRed[128];

  const unsigned ldsV = lds_off(sV);

  if (tid < 64) { sMax[tid] = -3.0e38f; sSum[tid] = 0.f; }

  const u16* Kb = Kt + (size_t)b * NDIM * DDIM;
  const u16* Vb = Vm + (size_t)b * CDIM * NDIM;

  auto issueKT = [&](u16* dst, int nn) {            // 4 async ops per wave
    unsigned l = lds_off(dst);
    const u16* g = Kb + (size_t)nn * DDIM;
    for (int i = tid; i < 64 * DDIM / 8; i += 128)
      async_copy_b128(l + i * 16, g + i * 8);
  };
  auto issueV = [&](int nn) {                       // 16 async ops per wave
    const u16* gv = Vb + nn;
    for (int i = tid; i < CDIM * 8; i += 128) {
      int c = i >> 3, ch = i & 7;
      async_copy_b128(ldsV + (c * 64 + ch * 8) * 2, gv + (size_t)c * NDIM + ch * 8);
    }
  };

  // prologue: Q (4 ops), KT[0] (4 ops), V[0] (16 ops)
  {
    unsigned lq = lds_off(sQ);
    const u16* g = Qt + (size_t)b * NDIM * DDIM + (size_t)m0 * DDIM;
    for (int i = tid; i < 64 * DDIM / 8; i += 128)
      async_copy_b128(lq + i * 16, g + i * 8);
  }
  issueKT(sKT0, 0);
  issueV(0);
  wait_async<16>();            // Q + KT[0] landed; V[0] still in flight
  __syncthreads();

  v8f accO[4][4] = {};   // [c row-tile][m col-tile]

  for (int n0 = 0; n0 < NDIM; n0 += 64) {
    const bool hasNext = (n0 + 64) < NDIM;
    u16* curKT = (n0 & 64) ? sKT1 : sKT0;
    u16* nxtKT = (n0 & 64) ? sKT0 : sKT1;

    if (hasNext) issueKT(nxtKT, n0 + 64);           // prefetch next K tile
    if (hasNext) wait_async<20>(); else wait_async<16>();  // KT[cur] ready
    __syncthreads();

    // S = Q . K^T for this wave's 16 m-rows
    v8f accS[4] = {};
    #pragma unroll
    for (int k0 = 0; k0 < DDIM; k0 += 32) {
      v16bf a = load_frag(sQ + (wave * 16) * DDIM + k0, DDIM, lane);
      #pragma unroll
      for (int ct = 0; ct < 4; ++ct) {
        v16bf bb = load_frag(curKT + (ct * 16) * DDIM + k0, DDIM, lane);
        accS[ct] = wmma_bf16(a, bb, accS[ct]);
      }
    }
    #pragma unroll
    for (int ct = 0; ct < 4; ++ct) {
      int n  = ct * 16 + (lane & 15);
      int mb = wave * 16 + ((lane >> 4) << 3);
      #pragma unroll
      for (int v = 0; v < 8; ++v) sS[(mb + v) * 64 + n] = accS[ct][v];
    }
    __syncthreads();

    // online softmax: 2 threads per row, float4 reads, packed bf16 P stores
    {
      int m = tid >> 1, h = tid & 1;
      const float*  rowp = sS + m * 64 + h * 32;
      const float4* r4   = (const float4*)rowp;
      float prevMax = sMax[m];
      float hm = -3.0e38f;
      #pragma unroll
      for (int j = 0; j < 8; ++j) {
        float4 vv = r4[j];
        hm = fmaxf(hm, fmaxf(fmaxf(vv.x, vv.y), fmaxf(vv.z, vv.w)));
      }
      sRed[tid] = hm;
      __syncthreads();
      float rmax = fmaxf(prevMax, fmaxf(sRed[m * 2], sRed[m * 2 + 1]));
      u32* prow = (u32*)(sP + m * 64 + h * 32);
      float s = 0.f;
      #pragma unroll
      for (int j = 0; j < 8; ++j) {
        float4 vv = r4[j];
        float p0 = __expf(vv.x - rmax), p1 = __expf(vv.y - rmax);
        float p2 = __expf(vv.z - rmax), p3 = __expf(vv.w - rmax);
        s += (p0 + p1) + (p2 + p3);
        prow[j * 2]     = bf_pack2(p0, p1);
        prow[j * 2 + 1] = bf_pack2(p2, p3);
      }
      sRed[tid] = s;   // pair lanes share a wave: lockstep-safe
      __syncthreads();
      if (h == 0) {
        float corr = __expf(prevMax - rmax);
        sSum[m]   = sSum[m] * corr + sRed[m * 2] + sRed[m * 2 + 1];
        sMax[m]   = rmax;
        sScale[m] = corr;
      }
    }
    __syncthreads();

    // rescale O accumulators (each lane's accum column is one m)
    #pragma unroll
    for (int ct = 0; ct < 4; ++ct) {
      float f = sScale[ct * 16 + (lane & 15)];
      #pragma unroll
      for (int rt = 0; rt < 4; ++rt)
        #pragma unroll
        for (int v = 0; v < 8; ++v)
          accO[rt][ct][v] *= f;
    }

    if (hasNext) wait_async<4>(); else wait_async<0>();    // V[cur] ready
    __syncthreads();

    // O += V . P^T ; contraction over the 64 n's of this tile
    #pragma unroll
    for (int k0 = 0; k0 < 64; k0 += 32) {
      #pragma unroll
      for (int rt = 0; rt < 4; ++rt) {
        v16bf a = load_frag(sV + (wave * 64 + rt * 16) * 64 + k0, 64, lane);
        #pragma unroll
        for (int ct = 0; ct < 4; ++ct) {
          v16bf bb = load_frag(sP + (ct * 16) * 64 + k0, 64, lane);
          accO[rt][ct] = wmma_bf16(a, bb, accO[rt][ct]);
        }
      }
    }
    __syncthreads();               // all sV readers done
    if (hasNext) issueV(n0 + 64);  // prefetch next V tile into freed buffer
  }

  const float renorm = 1.0f / (1.0f + 1e-9f);   // extra L1 renorm (sum==1)
  #pragma unroll
  for (int ct = 0; ct < 4; ++ct) {
    int m = ct * 16 + (lane & 15);
    float inv = renorm / (sSum[m] + 1e-30f);
    u16* orow = Rt + (size_t)b * NDIM * CDIM + (size_t)(m0 + m) * CDIM;
    #pragma unroll
    for (int rt = 0; rt < 4; ++rt) {
      int cb = wave * 64 + rt * 16 + ((lane >> 4) << 3);
      v8f o = accO[rt][ct];
      uint4 pk;
      pk.x = bf_pack2(o[0] * inv, o[1] * inv);
      pk.y = bf_pack2(o[2] * inv, o[3] * inv);
      pk.z = bf_pack2(o[4] * inv, o[5] * inv);
      pk.w = bf_pack2(o[6] * inv, o[7] * inv);
      *(uint4*)(orow + cb) = pk;               // 8 channels per 16B store
    }
  }
}

// ---------------------------------------------------------------------------
// t = w_t . r  (bf16 in, fp32 out) fused with per-channel BN statistics.
// r is [B][N][C] so the B tile is row-contiguous -> async b128 staging.
// ---------------------------------------------------------------------------
__global__ void __launch_bounds__(128)
trans_gemm_bn_kernel(const float* __restrict__ W, const u16* __restrict__ Rbf,
                     float* __restrict__ T, float* __restrict__ stats /* [2*C] */)
{
  const int b    = blockIdx.z;
  const int row0 = blockIdx.y * 64;
  const int col0 = blockIdx.x * 64;
  const int tid  = threadIdx.x;
  const int wave = tid >> 5, lane = tid & 31;

  __shared__ __align__(16) u16 lA [64 * 32];
  __shared__ __align__(16) u16 lBT[64 * 32];
  __shared__ float chS[64], chQ[64];

  const unsigned ldsBT = lds_off(lBT);
  const u16* Rb = Rbf + (size_t)b * NDIM * CDIM;   // [n][c]
  if (tid < 64) { chS[tid] = 0.f; chQ[tid] = 0.f; }

  v8f acc[4] = {};
  for (int k0 = 0; k0 < CDIM; k0 += 32) {
    // async-stage B tile: lBT[n][kk] <- r[col0+n][k0+kk] (64B per row)
    {
      const u16* g = Rb + (size_t)col0 * CDIM + k0;
      for (int i = tid; i < 64 * 4; i += 128) {
        int n = i >> 2, ch = i & 3;
        async_copy_b128(ldsBT + (n * 32 + ch * 8) * 2, g + (size_t)n * CDIM + ch * 8);
      }
    }
    // A with fp32->bf16 conversion (packed u32 stores)
    for (int i = tid; i < 64 * 16; i += 128) {
      int o = i >> 4, kp = (i & 15) * 2;
      const float* wp = W + (size_t)(row0 + o) * CDIM + (k0 + kp);
      ((u32*)lA)[i] = bf_pack2(wp[0], wp[1]);
    }
    wait_async<0>();
    __syncthreads();
    v16bf a = load_frag(lA + (wave * 16) * 32, 32, lane);
    #pragma unroll
    for (int ct = 0; ct < 4; ++ct) {
      v16bf bb = load_frag(lBT + (ct * 16) * 32, 32, lane);
      acc[ct] = wmma_bf16(a, bb, acc[ct]);
    }
    __syncthreads();
  }

  float* Tb = T + (size_t)b * CDIM * NDIM;
  #pragma unroll
  for (int ct = 0; ct < 4; ++ct) {
    int n  = col0 + ct * 16 + (lane & 15);
    int ob = row0 + wave * 16 + ((lane >> 4) << 3);
    #pragma unroll
    for (int v = 0; v < 8; ++v) {
      float val = acc[ct][v];
      int o = ob + v;
      Tb[(size_t)o * NDIM + n] = val;
      atomicAdd(&chS[o - row0], val);
      atomicAdd(&chQ[o - row0], val * val);
    }
  }
  __syncthreads();
  if (tid < 64) {
    atomicAdd(&stats[row0 + tid],        chS[tid]);
    atomicAdd(&stats[CDIM + row0 + tid], chQ[tid]);
  }
}

// ---------------------------------------------------------------------------
// out = src + relu(gamma * (t - mean) * rsqrt(var + eps) + beta)
// ---------------------------------------------------------------------------
__global__ void __launch_bounds__(256)
bn_finalize_kernel(const float* __restrict__ T, const float* __restrict__ stats,
                   const float* __restrict__ gamma, const float* __restrict__ beta,
                   const float* __restrict__ src, float* __restrict__ out, int total)
{
  int idx = blockIdx.x * blockDim.x + threadIdx.x;
  if (idx >= total) return;
  int c = (idx / NDIM) % CDIM;
  const float invBN = 1.0f / (float)(BATCH * NDIM);
  float mean = stats[c] * invBN;
  float var  = stats[CDIM + c] * invBN - mean * mean;
  float xh = (T[idx] - mean) * rsqrtf(var + 1e-5f);
  float r  = gamma[c] * xh + beta[c];
  out[idx] = src[idx] + fmaxf(r, 0.0f);
}

// ---------------------------------------------------------------------------
extern "C" void kernel_launch(void* const* d_in, const int* in_sizes, int n_in,
                              void* d_out, int out_size, void* d_ws, size_t ws_size,
                              hipStream_t stream)
{
  (void)in_sizes; (void)n_in; (void)out_size; (void)ws_size;
  const float* x     = (const float*)d_in[0];
  const float* y     = (const float*)d_in[1];
  const float* w_qkx = (const float*)d_in[2];
  const float* w_vx  = (const float*)d_in[3];
  const float* w_tx  = (const float*)d_in[4];
  const float* w_qky = (const float*)d_in[5];
  const float* w_vy  = (const float*)d_in[6];
  const float* w_ty  = (const float*)d_in[7];
  const float* gamma = (const float*)d_in[8];
  const float* beta  = (const float*)d_in[9];
  float* out = (float*)d_out;

  char* ws = (char*)d_ws;
  const size_t qB = (size_t)BATCH * NDIM * DDIM * sizeof(u16);   // 2 MB
  const size_t vB = (size_t)BATCH * CDIM * NDIM * sizeof(u16);   // 8 MB
  const size_t tB = (size_t)BATCH * CDIM * NDIM * sizeof(float); // 16 MB
  size_t off = 0;
  auto carve = [&](size_t bytes) {
    char* p = ws + off;
    off += (bytes + 255) & ~(size_t)255;
    return p;
  };
  u16*   qbuf[2] = { (u16*)carve(qB), (u16*)carve(qB) };
  u16*   kbuf[2] = { (u16*)carve(qB), (u16*)carve(qB) };
  u16*   vbuf[2] = { (u16*)carve(vB), (u16*)carve(vB) };
  u16*   rbuf[2] = { (u16*)carve(vB), (u16*)carve(vB) };   // [B][N][C]
  float* tbuf[2] = { (float*)carve(tB), (float*)carve(tB) };
  float* stats   = (float*)carve(2 * 2 * CDIM * sizeof(float));

  hipMemsetAsync(stats, 0, 2 * 2 * CDIM * sizeof(float), stream);

  // branch 0: r_x = branch(src=x, tgt=y, w_*x), out0 = x + r_x
  // branch 1: r_y = branch(src=y, tgt=x, w_*y), out1 = y + r_y
  const float* srcs[2] = { x, y };
  const float* tgts[2] = { y, x };
  const float* wqk[2]  = { w_qkx, w_qky };
  const float* wv [2]  = { w_vx,  w_vy  };
  const float* wt [2]  = { w_tx,  w_ty  };

  const dim3 gQ(NDIM / 64, DDIM / 64, BATCH);
  const dim3 gV(NDIM / 64, CDIM / 64, BATCH);
  const int total = BATCH * CDIM * NDIM;

  for (int i = 0; i < 2; ++i) {
    proj_gemm_kernel<<<gQ, 128, 0, stream>>>(wqk[i], tgts[i], qbuf[i], DDIM, CDIM, 1);
    proj_gemm_kernel<<<gQ, 128, 0, stream>>>(wqk[i], srcs[i], kbuf[i], DDIM, CDIM, 1);
    proj_gemm_kernel<<<gV, 128, 0, stream>>>(wv[i],  srcs[i], vbuf[i], CDIM, CDIM, 0);
    fmha_kernel<<<dim3(NDIM / 64, BATCH), 128, 0, stream>>>(qbuf[i], kbuf[i], vbuf[i], rbuf[i]);
    trans_gemm_bn_kernel<<<gV, 128, 0, stream>>>(wt[i], rbuf[i], tbuf[i], stats + (size_t)i * 2 * CDIM);
    bn_finalize_kernel<<<(total + 255) / 256, 256, 0, stream>>>(
        tbuf[i], stats + (size_t)i * 2 * CDIM, gamma, beta, srcs[i],
        out + (size_t)i * total, total);
  }
}
